// SetTransformer_82575041233736
// MI455X (gfx1250) — matrix operand order repaired and checked by reading, not compiled, not run on hardware
//
#include <hip/hip_runtime.h>
#include <hip/hip_bf16.h>
#include <math.h>

typedef __attribute__((ext_vector_type(16))) _Float16 v16h;
typedef __attribute__((ext_vector_type(8)))  _Float16 v8h;
typedef __attribute__((ext_vector_type(8)))  float    v8f;
typedef __attribute__((ext_vector_type(4)))  unsigned int v4u;
typedef __attribute__((ext_vector_type(8)))  int      v8i;
typedef __attribute__((ext_vector_type(4)))  int      v4i;

#define WMMA_F16(a,b,c) __builtin_amdgcn_wmma_f32_16x16x32_f16(false,(a),false,(b),(short)0,(c),false,false)

#if defined(__HIP_DEVICE_COMPILE__) && __has_builtin(__builtin_amdgcn_tensor_load_to_lds)
#define HAVE_TDM 1
#else
#define HAVE_TDM 0
#endif

#define Bsz    16
#define Lseq   2048
#define Fdim   64
#define Hh     8
#define DHd    64
#define Dm     512
#define FFd    2048
#define Kseeds 64

// ---------------------------------------------------------------------------
// WMMA fragments with natural-width loads (ISA 7.12.2):
//  A f16 16x32: lane row = lane&15, kh = (lane>>4)*8; halves hold k runs
//               {kh..kh+7} and {kh+16..kh+23} -> two contiguous b128 loads.
//  B f16 32x16: lane col = lane&15, kh = (lane>>4)*8; halves hold k runs
//               {kh..kh+7} and {kh+16..kh+23} -> two contiguous b128 loads when memory
//               is [n][k] (k fastest) -- i.e. pre-transposed weights.
//  C/D f32:     VGPR r -> M = r + 8*(lane>>4), N = lane&15.
// ---------------------------------------------------------------------------
__device__ __forceinline__ v16h frag_a_rm(const _Float16* p, int rs) {
  const int lane = threadIdx.x & 31;
  const _Float16* r0 = p + (lane & 15) * rs + ((lane >> 4) * 8);
  v8h lo = *(const v8h*)r0;
  v8h hi = *(const v8h*)(r0 + 16);
  return __builtin_shufflevector(lo, hi, 0,1,2,3,4,5,6,7,8,9,10,11,12,13,14,15);
}

__device__ __forceinline__ v16h frag_b_cm(const _Float16* p, int rs) {
  const int lane = threadIdx.x & 31;
  const _Float16* r0 = p + (size_t)(lane & 15) * rs + ((lane >> 4) * 8);
  v8h lo = *(const v8h*)r0;
  v8h hi = *(const v8h*)(r0 + 16);
  return __builtin_shufflevector(lo, hi, 0,1,2,3,4,5,6,7,8,9,10,11,12,13,14,15);
}

__device__ __forceinline__ v8f vzero8() {
  v8f z;
#pragma unroll
  for (int i = 0; i < 8; ++i) z[i] = 0.f;
  return z;
}

__device__ __forceinline__ float sigm(float x) { return 1.f / (1.f + __expf(-x)); }

// ---------------------------------------------------------------------------
// TDM: issue a 2D tensor_load_to_lds (fp16, tile rows x cols, row pitch in
// elements) and wait for TENSORcnt==0. Called by one wave; EXEC ignored by HW.
// ---------------------------------------------------------------------------
__device__ __forceinline__ void tdm_load_2d(unsigned int lds_off, const _Float16* gsrc,
                                            int rows, int cols, int pitch,
                                            int tdim0, int tdim1)
{
#if HAVE_TDM
  const unsigned long long ga = (unsigned long long)(uintptr_t)gsrc;
  v4u g0; v8i g1; v4i gz;
  g0[0] = 1u;                                   // count=1, user mode
  g0[1] = lds_off;                              // lds_addr [63:32]
  g0[2] = (unsigned int)ga;                     // global_addr[31:0]
  g0[3] = (unsigned int)((ga >> 32) & 0x01FFFFFFu) | (2u << 30);  // addr hi | type=2
  g1[0] = (int)(1u << 16);                      // data_size = 1 (2 bytes)
  g1[1] = (int)(((unsigned)tdim0 & 0xFFFFu) << 16);               // dim0 lo
  g1[2] = (int)((((unsigned)tdim0) >> 16) | (((unsigned)tdim1 & 0xFFFFu) << 16));
  g1[3] = (int)((((unsigned)tdim1) >> 16) | (((unsigned)cols & 0xFFFFu) << 16)); // tile_dim0
  g1[4] = (int)((unsigned)rows & 0xFFFFu);      // tile_dim1 (tile_dim2 = 0)
  g1[5] = (int)(unsigned)pitch;                 // tensor_dim0_stride lo
  g1[6] = 0; g1[7] = 0;
  gz[0] = gz[1] = gz[2] = gz[3] = 0;
#if defined(__clang_major__) && __clang_major__ >= 23
  v8i gz8; gz8[0]=gz8[1]=gz8[2]=gz8[3]=gz8[4]=gz8[5]=gz8[6]=gz8[7]=0;
  __builtin_amdgcn_tensor_load_to_lds(g0, g1, gz, gz, gz8, 0);
#else
  __builtin_amdgcn_tensor_load_to_lds(g0, g1, gz, gz, 0);
#endif
#if __has_builtin(__builtin_amdgcn_s_wait_tensorcnt)
  __builtin_amdgcn_s_wait_tensorcnt((short)0);
#else
  asm volatile("s_wait_tensorcnt 0x0" ::: "memory");
#endif
#else
  (void)lds_off; (void)gsrc; (void)rows; (void)cols; (void)pitch; (void)tdim0; (void)tdim1;
#endif
}

// ---------------------------------------------------------------------------
// GEMM: C[M,N] = act(A[M,K] @ BT^T + bias(+bias2)) (+ add). A fp16 row-major,
// BT fp16 [N,K] (pre-transposed weights, k contiguous -> direct global B
// fragments, L2-resident). A tile staged to LDS by TDM (wave 0) or fallback.
// M%32==0, N%64==0, K%32==0. 256 threads = 8 waves, 2x4 16x16 tiles.
// ---------------------------------------------------------------------------
__global__ __launch_bounds__(256) void k_gemm(
    const _Float16* __restrict__ A, const _Float16* __restrict__ BT,
    const float* __restrict__ bias, const float* __restrict__ bias2,
    const _Float16* __restrict__ add, _Float16* __restrict__ Ch,
    float* __restrict__ Cf, int M, int N, int Kd, int act)
{
  __shared__ _Float16 As[32 * 32];
  const int tid = threadIdx.x, lane = tid & 31, wid = tid >> 5;
  const int wm = wid >> 2, wn = wid & 3;
  const int row0 = blockIdx.y * 32, col0 = blockIdx.x * 64;
  v8f acc = vzero8();

  for (int k0 = 0; k0 < Kd; k0 += 32) {
    __syncthreads();                 // previous tile fully consumed
#if HAVE_TDM
    if (tid < 32)
      tdm_load_2d((unsigned int)(uintptr_t)As, A + (size_t)row0 * Kd + k0,
                  32, 32, Kd, Kd, M);
#else
    {
      const int r = tid >> 3, c = (tid & 7) * 4;
      const _Float16* ap = A + (size_t)(row0 + r) * Kd + k0 + c;
#pragma unroll
      for (int j = 0; j < 4; ++j) As[r * 32 + c + j] = ap[j];
    }
#endif
    __syncthreads();
    const _Float16* bp = BT + (size_t)(col0 + wn * 16 + (lane & 15)) * Kd + k0
                         + ((lane >> 4) * 16);
    __builtin_prefetch(bp + 32, 0, 1);   // next k-step weight fragment
    v16h af = frag_a_rm(As + wm * 16 * 32, 32);
    v16h bf = frag_b_cm(BT + (size_t)(col0 + wn * 16) * Kd + k0, Kd);
    acc = WMMA_F16(af, bf, acc);
  }

  const int n = col0 + wn * 16 + (lane & 15);
  float bs = bias ? bias[n] : 0.f;
  if (bias2) bs += bias2[n];
#pragma unroll
  for (int r = 0; r < 8; ++r) {
    const int m = row0 + wm * 16 + r + 8 * (lane >> 4);
    float v = acc[r] + bs;
    if (act == 1)      v = fmaxf(v, 0.f);
    else if (act == 2) v = (v > 0.f) ? v + 1.f : __expf(v);   // elu(x)+1
    const size_t idx = (size_t)m * N + n;
    if (add) v += (float)add[idx];
    if (Cf) Cf[idx] = v; else Ch[idx] = (_Float16)v;
  }
}

// ---------------------------------------------------------------------------
// LayerNorm over D=512 (fp16 in/out, fp32 math). One block per row.
// ---------------------------------------------------------------------------
__global__ __launch_bounds__(256) void k_ln(
    const _Float16* __restrict__ x, const float* __restrict__ g,
    const float* __restrict__ b, _Float16* __restrict__ y)
{
  __shared__ float red[256];
  const size_t row = blockIdx.x;
  const int t = threadIdx.x;
  const _Float16* xr = x + row * Dm;
  const float v0 = (float)xr[t], v1 = (float)xr[t + 256];
  red[t] = v0 + v1;
  __syncthreads();
  for (int o = 128; o > 0; o >>= 1) { if (t < o) red[t] += red[t + o]; __syncthreads(); }
  const float mean = red[0] * (1.f / (float)Dm);
  __syncthreads();
  const float d0 = v0 - mean, d1 = v1 - mean;
  red[t] = d0 * d0 + d1 * d1;
  __syncthreads();
  for (int o = 128; o > 0; o >>= 1) { if (t < o) red[t] += red[t + o]; __syncthreads(); }
  const float inv = rsqrtf(red[0] * (1.f / (float)Dm) + 1e-5f);
  y[row * Dm + t]       = (_Float16)(g[t]       * (d0 * inv) + b[t]);
  y[row * Dm + t + 256] = (_Float16)(g[t + 256] * (d1 * inv) + b[t + 256]);
}

// ksum[b,h,d] = sum_l phi_k[b,l,h,d]
__global__ void k_ksum(const _Float16* __restrict__ k, float* __restrict__ ks, int L)
{
  const int bh = blockIdx.x, b = bh / Hh, h = bh % Hh, d = threadIdx.x;
  const _Float16* p = k + (size_t)b * L * Dm + h * DHd + d;
  float s = 0.f;
  for (int l = 0; l < L; ++l) s += (float)p[(size_t)l * Dm];
  ks[(size_t)bh * DHd + d] = s;
}

// kv[b,h,d,m] = sum_l k[b,l,h,d]*v[b,l,h,m]; K/V tiles staged TRANSPOSED so
// both fragments are contiguous b128 LDS loads.
__global__ __launch_bounds__(256) void k_kv(
    const _Float16* __restrict__ k, const _Float16* __restrict__ v,
    float* __restrict__ kv, int L)
{
  __shared__ _Float16 Kt[64 * 40];   // [d][l], pitch 40
  __shared__ _Float16 Vt[64 * 40];   // [m][l]
  const int bh = blockIdx.x, b = bh / Hh, h = bh % Hh;
  const int tid = threadIdx.x, lane = tid & 31, wid = tid >> 5;
  const int wd  = wid >> 1;
  const int wmc = (wid & 1) * 32;
  const _Float16* kb = k + (size_t)b * L * Dm + h * DHd;
  const _Float16* vb = v + (size_t)b * L * Dm + h * DHd;
  v8f acc0 = vzero8(), acc1 = vzero8();
  for (int l0 = 0; l0 < L; l0 += 32) {
    const int r = tid >> 3, c = (tid & 7) * 8;
    const _Float16* kp = kb + (size_t)(l0 + r) * Dm + c;
    const _Float16* vp = vb + (size_t)(l0 + r) * Dm + c;
    __syncthreads();
#pragma unroll
    for (int j = 0; j < 8; ++j) { Kt[(c + j) * 40 + r] = kp[j]; Vt[(c + j) * 40 + r] = vp[j]; }
    __syncthreads();
    v16h af = frag_a_rm(Kt + wd * 16 * 40, 40);          // A[d][l]
    v16h b0 = frag_b_cm(Vt + wmc * 40, 40);              // B[l][m]
    v16h b1 = frag_b_cm(Vt + (wmc + 16) * 40, 40);
    acc0 = WMMA_F16(af, b0, acc0);
    acc1 = WMMA_F16(af, b1, acc1);
  }
  float* out = kv + (size_t)bh * DHd * DHd;
#pragma unroll
  for (int r = 0; r < 8; ++r) {
    const int d = wd * 16 + r + 8 * (lane >> 4);
    out[d * DHd + wmc + (lane & 15)]      = acc0[r];
    out[d * DHd + wmc + 16 + (lane & 15)] = acc1[r];
  }
}

// out[b,l,h,m] = z * sum_d q[b,l,h,d]*kv[b,h,d,m]; grid (B*H, L/32)
__global__ __launch_bounds__(256) void k_attn_out(
    const _Float16* __restrict__ q, const float* __restrict__ kv,
    const float* __restrict__ ksum, _Float16* __restrict__ out, int L)
{
  __shared__ _Float16 KVt[64 * 72];  // [m][d], pitch 72
  __shared__ _Float16 Qs[32 * 72];   // [l][d]
  __shared__ float zs[32];
  __shared__ float kss[64];
  const int bh = blockIdx.x, b = bh / Hh, h = bh % Hh;
  const int l0 = blockIdx.y * 32;
  const int tid = threadIdx.x, lane = tid & 31, wid = tid >> 5;
  const int wm = wid >> 2, wn = wid & 3;
  const float* kvb = kv + (size_t)bh * DHd * DHd;
  {
    const int m = tid >> 2, ds = (tid & 3) * 16;
#pragma unroll
    for (int j = 0; j < 16; ++j) KVt[m * 72 + ds + j] = (_Float16)kvb[(ds + j) * DHd + m];
  }
  if (tid < 64) kss[tid] = ksum[(size_t)bh * DHd + tid];
  const _Float16* qb = q + ((size_t)b * L + l0) * Dm + h * DHd;
  {
    const int r = tid >> 3, c = (tid & 7) * 8;
    const _Float16* qp = qb + (size_t)r * Dm + c;
#pragma unroll
    for (int j = 0; j < 8; ++j) Qs[r * 72 + c + j] = qp[j];
  }
  __syncthreads();
  if (tid < 32) {
    const _Float16* qr = qb + (size_t)tid * Dm;
    float s = 0.f;
    for (int d = 0; d < 64; ++d) s += (float)qr[d] * kss[d];
    zs[tid] = 1.f / (s + 1e-6f);
  }
  __syncthreads();
  v16h a0 = frag_a_rm(Qs + wm * 16 * 72, 72);
  v16h a1 = frag_a_rm(Qs + wm * 16 * 72 + 32, 72);
  v16h b0 = frag_b_cm(KVt + wn * 16 * 72, 72);
  v16h b1 = frag_b_cm(KVt + wn * 16 * 72 + 32, 72);
  v8f acc = vzero8();
  acc = WMMA_F16(a0, b0, acc);
  acc = WMMA_F16(a1, b1, acc);
  _Float16* ob = out + ((size_t)b * L + l0) * Dm + h * DHd;
#pragma unroll
  for (int r = 0; r < 8; ++r) {
    const int lrow = wm * 16 + r + 8 * (lane >> 4);
    ob[(size_t)lrow * Dm + wn * 16 + (lane & 15)] = (_Float16)(acc[r] * zs[lrow]);
  }
}

// ---------------------------------------------------------------------------
// Persistent LSTM: 1 workgroup, 16 waves. h[16][512] fp16 in LDS, c in VGPR
// accumulators. Whh fp16 [4D,512] (k contiguous) -> direct global B frags.
// ---------------------------------------------------------------------------
__global__ __launch_bounds__(512) void k_lstm(
    const _Float16* __restrict__ xg,       // [B, L, 4D]
    const _Float16* __restrict__ whh,      // [4D, D]
    _Float16* __restrict__ hs)             // [B, L, D]
{
  __shared__ _Float16 hsm[16 * Dm];
  const int tid = threadIdx.x, lane = tid & 31, wid = tid >> 5;   // wid 0..15
  for (int i = tid; i < 16 * Dm; i += 512) hsm[i] = (_Float16)0.f;
  v8f c[2];
  c[0] = vzero8(); c[1] = vzero8();
  float hnew[2][8];
  const int nb = lane & 15;
  const int rb = 8 * (lane >> 4);
  __syncthreads();

  for (int t = 0; t < Lseq; ++t) {
    v8f acc[4][2];
#pragma unroll
    for (int qg = 0; qg < 4; ++qg)
#pragma unroll
      for (int s = 0; s < 2; ++s) {
        const int n = qg * Dm + wid * 32 + s * 16 + nb;
#pragma unroll
        for (int r = 0; r < 8; ++r)
          acc[qg][s][r] = (float)xg[((size_t)(r + rb) * Lseq + t) * (4 * Dm) + n];
      }
    for (int k0 = 0; k0 < Dm; k0 += 32) {
      v16h af = frag_a_rm(hsm + k0, Dm);   // A[batch][k]
#pragma unroll
      for (int qg = 0; qg < 4; ++qg)
#pragma unroll
        for (int s = 0; s < 2; ++s) {
          v16h bf = frag_b_cm(whh + (size_t)(qg * Dm + wid * 32 + s * 16) * Dm + k0, Dm);
          acc[qg][s] = WMMA_F16(af, bf, acc[qg][s]);
        }
    }
#pragma unroll
    for (int s = 0; s < 2; ++s) {
      const int d = wid * 32 + s * 16 + nb;
#pragma unroll
      for (int r = 0; r < 8; ++r) {
        const float ig = sigm(acc[0][s][r]);
        const float fg = sigm(acc[1][s][r]);
        const float gg = tanhf(acc[2][s][r]);
        const float og = sigm(acc[3][s][r]);
        const float cc = fg * c[s][r] + ig * gg;
        c[s][r] = cc;
        const float hh = og * tanhf(cc);
        hnew[s][r] = hh;
        hs[((size_t)(r + rb) * Lseq + t) * Dm + d] = (_Float16)hh;
      }
    }
    __syncthreads();
#pragma unroll
    for (int s = 0; s < 2; ++s) {
      const int d = wid * 32 + s * 16 + nb;
#pragma unroll
      for (int r = 0; r < 8; ++r) hsm[(r + rb) * Dm + d] = (_Float16)hnew[s][r];
    }
    __syncthreads();
  }
}

// fp32 -> fp16 plain convert
__global__ void k_f2h(const float* __restrict__ a, _Float16* __restrict__ o, int n)
{
  const int i = blockIdx.x * 256 + threadIdx.x;
  if (i < n) o[i] = (_Float16)a[i];
}

// weight transpose-convert: w[K,N] fp32 -> o[N,K] fp16
__global__ void k_wT(const float* __restrict__ w, _Float16* __restrict__ o, int K, int N)
{
  const size_t i = (size_t)blockIdx.x * 256 + threadIdx.x;
  if (i >= (size_t)K * N) return;
  const int n = (int)(i / K), kk = (int)(i % K);
  o[i] = (_Float16)w[(size_t)kk * N + n];
}

__global__ void k_bcast_seeds(const float* __restrict__ s, _Float16* __restrict__ o)
{ // grid B*K, block 512
  const int bk = blockIdx.x, kk = bk % Kseeds;
  o[(size_t)bk * Dm + threadIdx.x] = (_Float16)s[(size_t)kk * Dm + threadIdx.x];
}

// ---------------------------------------------------------------------------
// Host orchestration
// ---------------------------------------------------------------------------
extern "C" void kernel_launch(void* const* d_in, const int* in_sizes, int n_in,
                              void* d_out, int out_size, void* d_ws, size_t ws_size,
                              hipStream_t stream)
{
  (void)in_sizes; (void)n_in; (void)out_size; (void)ws_size;
  auto F = [&](int i) -> const float* { return (const float*)d_in[i]; };

  // param index map (setup_inputs dict order, depth-first):
  // 0 src | 1,2 initial_ff | 3..6 lstm0 | 7..10 lstm1 | 11+16l enc_pre layers
  // 75,76 enc_pre norm | 77+16l enc_post | 141,142 norm | 143 seeds
  // 144..151 pool | 152,153 final_ff
  const int PRE = 11, PRE_N = 75, POST = 77, POST_N = 141;
  const int SEEDS = 143, POOL = 144, FINAL = 152;

  char* ws = (char*)d_ws;
  size_t off = 0;
  auto alloc = [&](size_t bytes) -> void* {
    void* p = ws + off; off += (bytes + 255) & ~(size_t)255; return p;
  };
  const size_t BL = (size_t)Bsz * Lseq;     // 32768
  const size_t BK = (size_t)Bsz * Kseeds;   // 1024

  _Float16* src16 = (_Float16*)alloc(BL * Fdim * 2);
  _Float16* xb    = (_Float16*)alloc(BL * Dm * 2);
  _Float16* tmp   = (_Float16*)alloc(BL * Dm * 2);
  _Float16* t2    = (_Float16*)alloc(BL * Dm * 2);
  _Float16* hid   = (_Float16*)alloc(BL * FFd * 2);   // FFN hidden / q,k,v / xg
  float*    kvb   = (float*)alloc((size_t)Bsz * Hh * DHd * DHd * 4);
  float*    ksb   = (float*)alloc((size_t)Bsz * Hh * DHd * 4);
  _Float16* pxb   = (_Float16*)alloc(BK * Dm * 2);
  _Float16* ptmp  = (_Float16*)alloc(BK * Dm * 2);
  _Float16* pt2   = (_Float16*)alloc(BK * Dm * 2);
  _Float16* phid  = (_Float16*)alloc(BK * FFd * 2);

  // fp16 weights: [N,K] (transposed) so B fragments read k contiguously
  auto wtr = [&](int idx, int K, int N) -> _Float16* {
    _Float16* p = (_Float16*)alloc((size_t)K * N * 2);
    k_wT<<<dim3((unsigned)(((size_t)K * N + 255) / 256)), 256, 0, stream>>>(F(idx), p, K, N);
    return p;
  };
  auto wcv = [&](int idx, size_t n) -> _Float16* {   // already [N,K]
    _Float16* p = (_Float16*)alloc(n * 2);
    k_f2h<<<dim3((unsigned)((n + 255) / 256)), 256, 0, stream>>>(F(idx), p, (int)n);
    return p;
  };

  _Float16* iniT = wtr(1, Fdim, Dm);
  _Float16* wih0 = wcv(3, (size_t)4 * Dm * Dm);
  _Float16* whh0 = wcv(4, (size_t)4 * Dm * Dm);
  _Float16* wih1 = wcv(7, (size_t)4 * Dm * Dm);
  _Float16* whh1 = wcv(8, (size_t)4 * Dm * Dm);
  struct LW { _Float16 *q, *k, *v, *o, *f1, *f2; };
  LW pre[4], post[4];
  for (int l = 0; l < 4; ++l) {
    int b = PRE + 16 * l;
    pre[l]  = { wtr(b+0,Dm,Dm), wtr(b+2,Dm,Dm), wtr(b+4,Dm,Dm), wtr(b+6,Dm,Dm),
                wtr(b+10,Dm,FFd), wtr(b+12,FFd,Dm) };
    b = POST + 16 * l;
    post[l] = { wtr(b+0,Dm,Dm), wtr(b+2,Dm,Dm), wtr(b+4,Dm,Dm), wtr(b+6,Dm,Dm),
                wtr(b+10,Dm,FFd), wtr(b+12,FFd,Dm) };
  }
  _Float16* poolW[4] = { wtr(POOL+0,Dm,Dm), wtr(POOL+2,Dm,Dm),
                         wtr(POOL+4,Dm,Dm), wtr(POOL+6,Dm,Dm) };
  _Float16* finT = wtr(FINAL, Dm, Fdim);

  auto gemm = [&](const _Float16* A, const _Float16* BT, const float* bias,
                  const float* bias2, const _Float16* add,
                  _Float16* Ch, float* Cf, int M, int N, int Kd, int act) {
    dim3 g(N / 64, M / 32);
    k_gemm<<<g, 256, 0, stream>>>(A, BT, bias, bias2, add, Ch, Cf, M, N, Kd, act);
  };

  auto enc_layer = [&](_Float16* x, int Lc, int Mc, int base, LW& W,
                       _Float16* qb_, _Float16* kb_, _Float16* vb_,
                       _Float16* ab_, _Float16* tb_, _Float16* hb_) {
    gemm(x, W.q, F(base+1), nullptr, nullptr, qb_, nullptr, Mc, Dm, Dm, 2);  // phi
    gemm(x, W.k, F(base+3), nullptr, nullptr, kb_, nullptr, Mc, Dm, Dm, 2);  // phi
    gemm(x, W.v, F(base+5), nullptr, nullptr, vb_, nullptr, Mc, Dm, Dm, 0);
    k_ksum<<<dim3(Bsz * Hh), 64, 0, stream>>>(kb_, ksb, Lc);
    k_kv<<<dim3(Bsz * Hh), 256, 0, stream>>>(kb_, vb_, kvb, Lc);
    k_attn_out<<<dim3(Bsz * Hh, Lc / 32), 256, 0, stream>>>(qb_, kvb, ksb, ab_, Lc);
    gemm(ab_, W.o, F(base+7), nullptr, x, tb_, nullptr, Mc, Dm, Dm, 0);      // +res
    k_ln<<<dim3(Mc), 256, 0, stream>>>(tb_, F(base+8), F(base+9), x);
    gemm(x, W.f1, F(base+11), nullptr, nullptr, hb_, nullptr, Mc, FFd, Dm, 1); // relu
    gemm(hb_, W.f2, F(base+13), nullptr, x, tb_, nullptr, Mc, Dm, FFd, 0);   // +res
    k_ln<<<dim3(Mc), 256, 0, stream>>>(tb_, F(base+14), F(base+15), x);
  };

  // 1) initial_ff
  k_f2h<<<dim3((unsigned)((BL * Fdim + 255) / 256)), 256, 0, stream>>>(
      F(0), src16, (int)(BL * Fdim));
  gemm(src16, iniT, F(2), nullptr, nullptr, xb, nullptr, (int)BL, Dm, Fdim, 0);

  // 2) LSTM x2: xg precompute (WMMA) then persistent recurrence
  gemm(xb,  wih0, F(5), F(6),  nullptr, hid, nullptr, (int)BL, 4 * Dm, Dm, 0);
  k_lstm<<<1, 512, 0, stream>>>(hid, whh0, tmp);
  gemm(tmp, wih1, F(9), F(10), nullptr, hid, nullptr, (int)BL, 4 * Dm, Dm, 0);
  k_lstm<<<1, 512, 0, stream>>>(hid, whh1, xb);

  // 3) enc_pre applied twice (weight-tied)
  _Float16* qb_ = hid;
  _Float16* kb_ = hid + BL * Dm;
  _Float16* vb_ = hid + 2 * BL * Dm;
  for (int rep = 0; rep < 2; ++rep) {
    for (int l = 0; l < 4; ++l)
      enc_layer(xb, Lseq, (int)BL, PRE + 16 * l, pre[l], qb_, kb_, vb_, tmp, t2, hid);
    k_ln<<<dim3((int)BL), 256, 0, stream>>>(xb, F(PRE_N), F(PRE_N + 1), xb);
  }

  // 4) seed pooling (no residual, no LN)
  k_bcast_seeds<<<dim3((int)BK), 512, 0, stream>>>(F(SEEDS), pt2);
  gemm(pt2, poolW[0], F(POOL+1), nullptr, nullptr, phid, nullptr, (int)BK, Dm, Dm, 2);
  gemm(xb,  poolW[1], F(POOL+3), nullptr, nullptr, kb_,  nullptr, (int)BL, Dm, Dm, 2);
  gemm(xb,  poolW[2], F(POOL+5), nullptr, nullptr, vb_,  nullptr, (int)BL, Dm, Dm, 0);
  k_ksum<<<dim3(Bsz * Hh), 64, 0, stream>>>(kb_, ksb, Lseq);
  k_kv<<<dim3(Bsz * Hh), 256, 0, stream>>>(kb_, vb_, kvb, Lseq);
  k_attn_out<<<dim3(Bsz * Hh, Kseeds / 32), 256, 0, stream>>>(phid, kvb, ksb, ptmp, Kseeds);
  gemm(ptmp, poolW[3], F(POOL+7), nullptr, nullptr, pxb, nullptr, (int)BK, Dm, Dm, 0);

  // 5) enc_post applied twice on [B,K,D]
  _Float16* pqb = hid;
  _Float16* pkb = hid + BK * Dm;
  _Float16* pvb = hid + 2 * BK * Dm;
  for (int rep = 0; rep < 2; ++rep) {
    for (int l = 0; l < 4; ++l)
      enc_layer(pxb, Kseeds, (int)BK, POST + 16 * l, post[l], pqb, pkb, pvb, ptmp, pt2, phid);
    k_ln<<<dim3((int)BK), 256, 0, stream>>>(pxb, F(POST_N), F(POST_N + 1), pxb);
  }

  // 6) final_ff -> fp32 d_out
  gemm(pxb, finT, F(FINAL + 1), nullptr, nullptr, nullptr, (float*)d_out,
       (int)BK, Fdim, Dm, 0);
}
